// ParallelTransformerBlock_79336635892070
// MI455X (gfx1250) — matrix operand order repaired and checked by reading, not compiled, not run on hardware
//
#include <hip/hip_runtime.h>
#include <cstdint>

// ---------------- problem constants ----------------
constexpr int BATCH = 2;
constexpr int SEQ   = 1024;
constexpr int DIM   = 2048;
constexpr int DH    = 128;
constexpr int HEADS = 16;
constexpr int FF    = 8192;
constexpr int NTOT  = DIM + DH + DH + FF + FF;   // 18688
constexpr int MROWS = BATCH * SEQ;               // 2048
constexpr int KCOL0 = DIM;                       // k columns in fused proj
constexpr int VCOL0 = DIM + DH;
constexpr int FCOL0 = DIM + 2 * DH;              // ff columns
constexpr int GCOL0 = DIM + 2 * DH + FF;         // gate columns

// ---------------- workspace layout (bytes, all 256B aligned) ----------------
constexpr size_t SZ_WI   = (size_t)DIM * NTOT * 2;
constexpr size_t SZ_AWO  = (size_t)DIM * DIM  * 2;
constexpr size_t SZ_FWO  = (size_t)FF  * DIM  * 2;
constexpr size_t SZ_XN   = (size_t)MROWS * DIM * 2;
constexpr size_t SZ_PROJ = (size_t)MROWS * NTOT * 2;
constexpr size_t SZ_Q    = (size_t)BATCH * HEADS * SEQ * DH * 2;
constexpr size_t SZ_KV   = (size_t)BATCH * SEQ * DH * 2;
constexpr size_t SZ_AO   = (size_t)MROWS * DIM * 2;

constexpr size_t OFF_WI   = 0;
constexpr size_t OFF_AWO  = OFF_WI   + SZ_WI;
constexpr size_t OFF_FWO  = OFF_AWO  + SZ_AWO;
constexpr size_t OFF_XN   = OFF_FWO  + SZ_FWO;
constexpr size_t OFF_PROJ = OFF_XN   + SZ_XN;
constexpr size_t OFF_Q    = OFF_PROJ + SZ_PROJ;
constexpr size_t OFF_K    = OFF_Q    + SZ_Q;
constexpr size_t OFF_V    = OFF_K    + SZ_KV;
constexpr size_t OFF_AO   = OFF_V    + SZ_KV;
constexpr size_t WS_NEEDED = OFF_AO + SZ_AO;

// ---------------- types ----------------
typedef __attribute__((ext_vector_type(16))) __bf16 v16bf;
typedef __attribute__((ext_vector_type(8)))  float  v8f;
typedef __attribute__((ext_vector_type(4)))  unsigned int u32x4;
typedef __attribute__((ext_vector_type(8)))  int          i32x8;
typedef __attribute__((ext_vector_type(4)))  int          i32x4;

union Frag { v16bf v; uint4 u[2]; };

__device__ __forceinline__ v8f wmma_bf16(const Frag& a, const Frag& b, v8f c) {
    // D(16x16,f32) = A(16x32,bf16) x B(32x16,bf16) + C
    return __builtin_amdgcn_wmma_f32_16x16x32_bf16(false, a.v, false, b.v,
                                                   (short)0, c, false, false);
}

__device__ __forceinline__ v8f v8f_zero() {
    v8f z = {0.f,0.f,0.f,0.f,0.f,0.f,0.f,0.f};
    return z;
}

// ---------------- TDM: 2D tile global->LDS (data_size = 2B, bf16) -----------
// LDS dest has rows of 32 halfs (16 DWORDs) padded by 8 halfs (4 DWORDs),
// matching pad_interval code 3 (16 DW) / pad_amount code 3 (4 DW).
// This toolchain exposes the 6-arg builtin:
//   (uint32x4 g0, int32x8 g1, int32x4 g2, int32x4 g3, int32x8 gx, i32 cpol)
__device__ __forceinline__ void tdm_load_tile_2d(unsigned ldsAddr, const void* gaddr,
                                                 unsigned tensorW, unsigned tensorH,
                                                 unsigned tileW,   unsigned tileH,
                                                 unsigned long long strideElems)
{
    unsigned long long ga = (unsigned long long)(uintptr_t)gaddr;
    u32x4 g0;
    g0[0] = 1u;                                        // count = 1 valid descriptor
    g0[1] = ldsAddr;                                   // LDS byte address
    g0[2] = (unsigned)(ga & 0xFFFFFFFFu);              // global addr [31:0]
    g0[3] = (unsigned)((ga >> 32) & 0x01FFFFFFu)       // global addr [56:32]
          | (2u << 30);                                // type = 2 ("image")
    i32x8 g1;
    g1[0] = (int)((1u << 16)                           // data_size = 2 bytes
                | (1u << 20)                           // pad_enable
                | (3u << 22)                           // pad_interval = 16 DWORDs
                | (3u << 25));                         // pad_amount   = 4 DWORDs
    g1[1] = (int)((tensorW & 0xFFFFu) << 16);          // tensor_dim0 [15:0]
    g1[2] = (int)((tensorW >> 16) | ((tensorH & 0xFFFFu) << 16));
    g1[3] = (int)((tensorH >> 16) | (tileW << 16));    // tile_dim0
    g1[4] = (int)(tileH & 0xFFFFu);                    // tile_dim1 (tile_dim2 = 0)
    g1[5] = (int)(strideElems & 0xFFFFFFFFull);        // tensor_dim0_stride lo
    g1[6] = (int)((strideElems >> 32) & 0xFFFFull);    // stride hi (dim1_stride = 0)
    g1[7] = 0;
    i32x4 z4 = {0, 0, 0, 0};
    i32x8 z8 = {0, 0, 0, 0, 0, 0, 0, 0};
    __builtin_amdgcn_tensor_load_to_lds(g0, g1, z4, z4, z8, 0);
}

// ---------------- kernel 1: fp32 -> bf16 convert ----------------
__global__ __launch_bounds__(256) void cvt_bf16_kernel(const float* __restrict__ in,
                                                       __bf16* __restrict__ out,
                                                       long long n)
{
    long long i = (long long)blockIdx.x * blockDim.x + threadIdx.x;
    long long stride = (long long)gridDim.x * blockDim.x;
    for (; i < n; i += stride) out[i] = (__bf16)in[i];
}

// ---------------- kernel 2: LayerNorm -> bf16 ----------------
__global__ __launch_bounds__(256) void ln_kernel(const float* __restrict__ x,
                                                 const float* __restrict__ gamma,
                                                 __bf16* __restrict__ xn)
{
    const int row = blockIdx.x;
    const int tid = threadIdx.x;
    const float* xr = x + (size_t)row * DIM;
    float s = 0.f, s2 = 0.f;
    for (int i = tid; i < DIM; i += 256) { float v = xr[i]; s += v; s2 += v * v; }
    __shared__ float rs[256], rs2[256];
    rs[tid] = s; rs2[tid] = s2;
    __syncthreads();
    for (int o = 128; o > 0; o >>= 1) {
        if (tid < o) { rs[tid] += rs[tid + o]; rs2[tid] += rs2[tid + o]; }
        __syncthreads();
    }
    const float mean = rs[0] * (1.f / DIM);
    const float var  = rs2[0] * (1.f / DIM) - mean * mean;
    const float rsq  = rsqrtf(var + 1e-5f);
    __bf16* xo = xn + (size_t)row * DIM;
    for (int i = tid; i < DIM; i += 256)
        xo[i] = (__bf16)((xr[i] - mean) * rsq * gamma[i]);
}

// ---------------- kernel 3: WMMA bf16 GEMM  C[M x N] = A[M x K] * W[K x N] ---
// block tile 128x128, K-step 32.  8 waves arranged 2(M) x 4(N); each wave
// computes a 64x32 tile = 4x2 WMMA 16x16 tiles.
// EPI: 0 = store bf16, 1 = store f32, 2 = accumulate f32.
template<int EPI>
__global__ __launch_bounds__(256) void gemm_bf16_kernel(
    const __bf16* __restrict__ A, int lda,
    const __bf16* __restrict__ W, int ldw,
    void* __restrict__ C, int ldc, int K)
{
    __shared__ __align__(16) __bf16 As[128][40];   // [m][k], +8 halfs pad
    __shared__ __align__(16) __bf16 Bs[128][40];   // transposed: [n][k]

    const int n0 = blockIdx.x * 128;
    const int m0 = blockIdx.y * 128;
    const int tid  = threadIdx.x;
    const int wave = tid >> 5, lane = tid & 31;
    const int wm = wave & 1, wn = wave >> 1;       // 2 x 4 wave grid
    const int r  = lane & 15, hi = lane >> 4;

    v8f acc[4][2];
    #pragma unroll
    for (int i = 0; i < 4; ++i)
        #pragma unroll
        for (int j = 0; j < 2; ++j) acc[i][j] = v8f_zero();

    const unsigned ldsA = (unsigned)(uintptr_t)(&As[0][0]);  // low 32b = LDS offset

    for (int k0 = 0; k0 < K; k0 += 32) {
        __syncthreads();   // previous iteration's reads done before refill
        // A tile (128 x 32) via Tensor Data Mover, issued once by wave 0
        if (wave == 0)
            tdm_load_tile_2d(ldsA, A + (size_t)m0 * lda + k0,
                             (unsigned)K, (unsigned)MROWS, 32u, 128u,
                             (unsigned long long)lda);
        // B tile (32 x 128) loaded coalesced and stored transposed -> Bs[n][k]
        {
            const int kr  = tid >> 3;              // 0..31
            const int nc0 = (tid & 7) << 4;        // 0,16,...,112
            const __bf16* gp = W + (size_t)(k0 + kr) * ldw + (n0 + nc0);
            union { uint4 u[2]; __bf16 h[16]; } t;
            t.u[0] = *(const uint4*)gp;
            t.u[1] = *(const uint4*)(gp + 8);
            #pragma unroll
            for (int i = 0; i < 16; ++i) Bs[nc0 + i][kr] = t.h[i];
        }
        if (wave == 0) __builtin_amdgcn_s_wait_tensorcnt(0);
        __syncthreads();

        // A fragments: lane r = row; K halves 0..7/16..23 (lanes<16) or 8..15/24..31
        Frag af[4];
        #pragma unroll
        for (int mt = 0; mt < 4; ++mt) {
            const __bf16* p = &As[wm * 64 + mt * 16 + r][hi * 8];
            af[mt].u[0] = *(const uint4*)p;
            af[mt].u[1] = *(const uint4*)(p + 16);
        }
        #pragma unroll
        for (int nt = 0; nt < 2; ++nt) {
            // B fragment: lane r = col; K halves 0..7/16..23 (lanes<16) or 8..15/24..31
            Frag bfr;
            const __bf16* p = &Bs[wn * 32 + nt * 16 + r][hi * 8];
            bfr.u[0] = *(const uint4*)p;
            bfr.u[1] = *(const uint4*)(p + 16);
            #pragma unroll
            for (int mt = 0; mt < 4; ++mt)
                acc[mt][nt] = wmma_bf16(af[mt], bfr, acc[mt][nt]);
        }
    }

    // epilogue: C layout -> lane col = r, row = elem + 8*hi
    #pragma unroll
    for (int mt = 0; mt < 4; ++mt)
        #pragma unroll
        for (int nt = 0; nt < 2; ++nt) {
            const int gr0 = m0 + wm * 64 + mt * 16 + hi * 8;
            const int gc  = n0 + wn * 32 + nt * 16 + r;
            #pragma unroll
            for (int e = 0; e < 8; ++e) {
                const size_t idx = (size_t)(gr0 + e) * ldc + gc;
                const float val = acc[mt][nt][e];
                if (EPI == 0)      ((__bf16*)C)[idx] = (__bf16)val;
                else if (EPI == 1) ((float*)C)[idx]  = val;
                else               ((float*)C)[idx] += val;
            }
        }
}

// ---------------- kernel 4: rope/scale/silu fusion ----------------
__global__ __launch_bounds__(256) void fuse_kernel(
    __bf16* __restrict__ proj,
    const float* __restrict__ sinp, const float* __restrict__ cosp,
    __bf16* __restrict__ qo, __bf16* __restrict__ ko, __bf16* __restrict__ vo)
{
    const int row = blockIdx.x;            // b*SEQ + n
    const int b = row / SEQ, n = row % SEQ;
    const int tid = threadIdx.x;
    __bf16* pr = proj + (size_t)row * NTOT;
    const float* sn = sinp + (size_t)n * DH;
    const float* cn = cosp + (size_t)n * DH;
    const float scale = 0.08838834764831845f;   // 128^-0.5

    // q: scale + rope, scatter to (b, h, n, d)
    for (int pi = tid; pi < DIM / 2; pi += 256) {
        const int e = 2 * pi;
        const int h = e >> 7, d = e & 127;
        const float qe = (float)pr[e]     * scale;
        const float qd = (float)pr[e + 1] * scale;
        const float re = qe * cn[d]     - qd * sn[d];
        const float ro = qd * cn[d + 1] + qe * sn[d + 1];
        const size_t base = (((size_t)(b * HEADS + h)) * SEQ + n) * DH + d;
        qo[base]     = (__bf16)re;
        qo[base + 1] = (__bf16)ro;
    }
    // k: rope (multi-query, one head)
    if (tid < DH / 2) {
        const int d = 2 * tid;
        const float ke = (float)pr[KCOL0 + d];
        const float kd = (float)pr[KCOL0 + d + 1];
        const size_t base = (size_t)row * DH + d;
        ko[base]     = (__bf16)(ke * cn[d]     - kd * sn[d]);
        ko[base + 1] = (__bf16)(kd * cn[d + 1] + ke * sn[d + 1]);
    }
    // v: copy
    if (tid < DH) vo[(size_t)row * DH + tid] = pr[VCOL0 + tid];
    // ff * silu(gate), in place over the ff columns
    for (int i = tid; i < FF; i += 256) {
        const float f = (float)pr[FCOL0 + i];
        const float g = (float)pr[GCOL0 + i];
        pr[FCOL0 + i] = (__bf16)(f * g / (1.f + __expf(-g)));
    }
}

// ---------------- kernel 5: causal flash attention (multi-query) -------------
// grid = B*H*(SEQ/128); 8 waves, each owns 16 q rows; 32 keys per step.
__global__ __launch_bounds__(256) void attn_kernel(
    const __bf16* __restrict__ Q,   // (B*H, SEQ, 128)
    const __bf16* __restrict__ Kc,  // (B, SEQ, 128)
    const __bf16* __restrict__ Vc,  // (B, SEQ, 128)
    __bf16* __restrict__ AO)        // (B, SEQ, HEADS*128)
{
    __shared__ __align__(16) __bf16 Vt[128][40];      // V tile transposed [d][key]
    __shared__ __align__(16) __bf16 Pl[8][16][40];    // per-wave P staging

    const int qb = blockIdx.x & 7;
    const int bh = blockIdx.x >> 3;
    const int b  = bh >> 4, h = bh & 15;
    const int tid = threadIdx.x, wave = tid >> 5, lane = tid & 31;
    const int r = lane & 15, hi = lane >> 4;
    const int qrow0 = qb * 128 + wave * 16;

    // Q fragments (16 rows x 128 dims) resident in registers
    Frag qf[4];
    const __bf16* qbase = Q + (((size_t)bh) * SEQ + qrow0) * DH;
    #pragma unroll
    for (int c = 0; c < 4; ++c) {
        const __bf16* p = qbase + (size_t)r * DH + c * 32 + hi * 8;
        qf[c].u[0] = *(const uint4*)p;
        qf[c].u[1] = *(const uint4*)(p + 16);
    }

    float m[8], l[8];
    v8f o[8];
    #pragma unroll
    for (int e = 0; e < 8; ++e) { m[e] = -3.0e38f; l[e] = 0.f; }
    #pragma unroll
    for (int nt = 0; nt < 8; ++nt) o[nt] = v8f_zero();

    const int kbend = (qb + 1) * 4;
    for (int kb = 0; kb < kbend; ++kb) {
        const int j0 = kb * 32;
        __syncthreads();
        // stage V tile (32 keys x 128 dims) transposed into LDS
        {
            const int key = tid >> 3;
            const int d0  = (tid & 7) << 4;
            const __bf16* vp = Vc + (((size_t)b) * SEQ + j0 + key) * DH + d0;
            union { uint4 u[2]; __bf16 hh[16]; } t;
            t.u[0] = *(const uint4*)vp;
            t.u[1] = *(const uint4*)(vp + 8);
            #pragma unroll
            for (int i = 0; i < 16; ++i) Vt[d0 + i][key] = t.hh[i];
        }
        __syncthreads();

        // sim tiles: S[jt] (16 q x 16 keys), K fragments loaded direct (contiguous)
        v8f s[2];
        s[0] = v8f_zero(); s[1] = v8f_zero();
        #pragma unroll
        for (int jt = 0; jt < 2; ++jt) {
            #pragma unroll
            for (int c = 0; c < 4; ++c) {
                Frag kf;
                const __bf16* kp = Kc + (((size_t)b) * SEQ + j0 + jt * 16 + r) * DH
                                 + c * 32 + hi * 8;
                kf.u[0] = *(const uint4*)kp;
                kf.u[1] = *(const uint4*)(kp + 16);
                s[jt] = wmma_bf16(qf[c], kf, s[jt]);
            }
        }

        // causal mask + online softmax (row = e + 8*hi, cols = j0+r / j0+16+r)
        float rmax[8];
        #pragma unroll
        for (int e = 0; e < 8; ++e) {
            const int row = qrow0 + e + 8 * hi;
            float a0 = s[0][e], a1 = s[1][e];
            if (j0 + r > row)      a0 = -1e30f;
            if (j0 + 16 + r > row) a1 = -1e30f;
            s[0][e] = a0; s[1][e] = a1;
            rmax[e] = fmaxf(a0, a1);
        }
        #pragma unroll
        for (int off = 1; off < 16; off <<= 1)
            #pragma unroll
            for (int e = 0; e < 8; ++e)
                rmax[e] = fmaxf(rmax[e], __shfl_xor(rmax[e], off, 32));

        float rsum[8];
        #pragma unroll
        for (int e = 0; e < 8; ++e) {
            const float mn = fmaxf(m[e], rmax[e]);
            const float sc = __expf(m[e] - mn);
            m[e] = mn;
            const float p0 = __expf(s[0][e] - mn);
            const float p1 = __expf(s[1][e] - mn);
            s[0][e] = p0; s[1][e] = p1;
            rsum[e] = p0 + p1;
            l[e] *= sc;
            #pragma unroll
            for (int nt = 0; nt < 8; ++nt) o[nt][e] *= sc;
        }
        #pragma unroll
        for (int off = 1; off < 16; off <<= 1)
            #pragma unroll
            for (int e = 0; e < 8; ++e)
                rsum[e] += __shfl_xor(rsum[e], off, 32);
        #pragma unroll
        for (int e = 0; e < 8; ++e) l[e] += rsum[e];

        // transpose P (C layout -> A layout) through per-wave LDS tile
        #pragma unroll
        for (int e = 0; e < 8; ++e) {
            Pl[wave][e + 8 * hi][r]      = (__bf16)s[0][e];
            Pl[wave][e + 8 * hi][16 + r] = (__bf16)s[1][e];
        }
        Frag pf;
        {
            const __bf16* pp = &Pl[wave][r][hi * 8];
            pf.u[0] = *(const uint4*)pp;
            pf.u[1] = *(const uint4*)(pp + 16);
        }
        // O += P(16x32) @ V(32x128)
        #pragma unroll
        for (int nt = 0; nt < 8; ++nt) {
            Frag vf;
            const __bf16* vp = &Vt[nt * 16 + r][hi * 8];
            vf.u[0] = *(const uint4*)vp;
            vf.u[1] = *(const uint4*)(vp + 16);
            o[nt] = wmma_bf16(pf, vf, o[nt]);
        }
    }

    // normalize and scatter to (b, n, h*128 + d)
    float inv[8];
    #pragma unroll
    for (int e = 0; e < 8; ++e) inv[e] = (l[e] > 0.f) ? 1.f / l[e] : 0.f;
    #pragma unroll
    for (int nt = 0; nt < 8; ++nt)
        #pragma unroll
        for (int e = 0; e < 8; ++e) {
            const int row = qrow0 + e + 8 * hi;
            AO[(((size_t)b) * SEQ + row) * DIM + h * DH + nt * 16 + r]
                = (__bf16)(o[nt][e] * inv[e]);
        }
}

// ---------------- host launcher ----------------
extern "C" void kernel_launch(void* const* d_in, const int* in_sizes, int n_in,
                              void* d_out, int out_size, void* d_ws, size_t ws_size,
                              hipStream_t stream)
{
    (void)in_sizes; (void)n_in; (void)out_size;
    if (ws_size < WS_NEEDED) return;

    const float* x     = (const float*)d_in[0];
    const float* gamma = (const float*)d_in[1];
    const float* wi    = (const float*)d_in[2];
    const float* awo   = (const float*)d_in[3];
    const float* fwo   = (const float*)d_in[4];
    const float* sinp  = (const float*)d_in[5];
    const float* cosp  = (const float*)d_in[6];
    // d_in[7] = causal mask (recomputed analytically in-kernel)

    char* ws = (char*)d_ws;
    __bf16* wi_b   = (__bf16*)(ws + OFF_WI);
    __bf16* awo_b  = (__bf16*)(ws + OFF_AWO);
    __bf16* fwo_b  = (__bf16*)(ws + OFF_FWO);
    __bf16* xn_b   = (__bf16*)(ws + OFF_XN);
    __bf16* proj_b = (__bf16*)(ws + OFF_PROJ);
    __bf16* q_b    = (__bf16*)(ws + OFF_Q);
    __bf16* k_b    = (__bf16*)(ws + OFF_K);
    __bf16* v_b    = (__bf16*)(ws + OFF_V);
    __bf16* ao_b   = (__bf16*)(ws + OFF_AO);
    float*  out    = (float*)d_out;

    const dim3 blk(256);

    cvt_bf16_kernel<<<4096, blk, 0, stream>>>(wi,  wi_b,  (long long)DIM * NTOT);
    cvt_bf16_kernel<<<1024, blk, 0, stream>>>(awo, awo_b, (long long)DIM * DIM);
    cvt_bf16_kernel<<<2048, blk, 0, stream>>>(fwo, fwo_b, (long long)FF * DIM);

    ln_kernel<<<MROWS, blk, 0, stream>>>(x, gamma, xn_b);

    // fused projection: (2048 x 2048) @ (2048 x 18688) -> bf16 proj
    gemm_bf16_kernel<0><<<dim3(NTOT / 128, MROWS / 128), blk, 0, stream>>>(
        xn_b, DIM, wi_b, NTOT, (void*)proj_b, NTOT, DIM);

    fuse_kernel<<<MROWS, blk, 0, stream>>>(proj_b, sinp, cosp, q_b, k_b, v_b);

    attn_kernel<<<BATCH * HEADS * (SEQ / 128), blk, 0, stream>>>(q_b, k_b, v_b, ao_b);

    // attn_out @ attn_wo -> f32 store
    gemm_bf16_kernel<1><<<dim3(DIM / 128, MROWS / 128), blk, 0, stream>>>(
        ao_b, DIM, awo_b, DIM, (void*)out, DIM, DIM);

    // (ff * silu(gate)) @ ff_wo -> f32 accumulate (ffact lives in proj, lda = NTOT)
    gemm_bf16_kernel<2><<<dim3(DIM / 128, MROWS / 128), blk, 0, stream>>>(
        proj_b + FCOL0, NTOT, fwo_b, DIM, (void*)out, DIM, FF);
}
